// RelativeMultiheadAttention_8959301779997
// MI455X (gfx1250) — matrix operand order repaired and checked by reading, not compiled, not run on hardware
//
#include <hip/hip_runtime.h>

typedef __attribute__((ext_vector_type(16))) _Float16 v16h;
typedef __attribute__((ext_vector_type(8)))  _Float16 v8h;
typedef __attribute__((ext_vector_type(4)))  _Float16 v4h;
typedef __attribute__((ext_vector_type(8)))  float    v8f;

#define EMBED   1024
#define HEADS   16
#define HEAD_DIM 64
#define MAX_REL 16
#define BATCH   4
#define SEQ     2048
#define ROWS    (BATCH * SEQ)                 // 8192
#define NBIAS   ((2 * MAX_REL + 1) * HEADS)   // 528

union FragAB { v16h v; v8h h[2]; };

__device__ __forceinline__ v8f wmma_f16(v16h a, v16h b, v8f c) {
  return __builtin_amdgcn_wmma_f32_16x16x32_f16(false, a, false, b,
                                                (short)0, c, false, false);
}

__device__ __forceinline__ void lds_fence() {
  __builtin_amdgcn_wave_barrier();
  asm volatile("s_wait_dscnt 0" ::: "memory");
  __builtin_amdgcn_wave_barrier();
}

// ---------------- stage 1a: fp32 -> fp16 for x ----------------
__global__ void cvt_x_kernel(const float* __restrict__ x,
                             _Float16* __restrict__ xh) {
  size_t i = ((size_t)blockIdx.x * blockDim.x + threadIdx.x) * 4;
  float4 f = *(const float4*)(x + i);
  v4h o;
  o.x = (_Float16)f.x; o.y = (_Float16)f.y;
  o.z = (_Float16)f.z; o.w = (_Float16)f.w;
  *(v4h*)(xh + i) = o;
}

// ---------------- stage 1b: transpose + convert W[k][n] -> Wt[n][k] ----------
__global__ void cvt_wT_kernel(const float* __restrict__ W,
                              _Float16* __restrict__ Wt) {
  __shared__ float t[32][33];
  const int kb = blockIdx.x * 32, nb = blockIdx.y * 32;
  const int tx = threadIdx.x, ty = threadIdx.y;        // 32 x 8
#pragma unroll
  for (int i = 0; i < 4; ++i)
    t[ty + i * 8][tx] = W[(size_t)(kb + ty + i * 8) * EMBED + nb + tx];
  __syncthreads();
#pragma unroll
  for (int i = 0; i < 4; ++i)
    Wt[(size_t)(nb + ty + i * 8) * EMBED + kb + tx] =
        (_Float16)t[tx][ty + i * 8];
}

// ---------------- stage 2: projection GEMM  Y = (Xh @ W + b) * scale --------
// Xh: [8192,1024] f16 row-major.  Wt: [N=1024][K=1024] f16 (pre-transposed).
// vmode==0: out[(b,h,s,d)] (Q,K layout).  vmode==1: out[(b,h,d,s)] (V^T).
// Wave tile 64x32 (4 M-tiles x 2 N-tiles); block = 8 waves as 2M x 4N = 128x128.
__global__ __launch_bounds__(256) void gemm_proj_kernel(
    const _Float16* __restrict__ Xh, const _Float16* __restrict__ Wt,
    const float* __restrict__ bias, _Float16* __restrict__ Out, int vmode,
    float scale) {
  const int wave = threadIdx.x >> 5;
  const int lane = threadIdx.x & 31;
  const int lh = lane & 15;
  const int lg = lane >> 4;                  // 0 | 1
  const int bm = (blockIdx.x & 63) * 128;    // 64 M-blocks
  const int bn = (blockIdx.x >> 6) * 128;    // 8 N-blocks
  const int wm = bm + (wave >> 2) * 64;      // 2 waves down (64 rows each)
  const int wn = bn + (wave & 3) * 32;       // 4 waves across (32 cols each)

  v8f c[4][2];
#pragma unroll
  for (int i = 0; i < 4; ++i)
#pragma unroll
    for (int j = 0; j < 2; ++j) c[i][j] = {};

  for (int ks = 0; ks < EMBED / 32; ++ks) {
    FragAB a[4], bf[2];
#pragma unroll
    for (int mi = 0; mi < 4; ++mi) {
      const _Float16* xr =
          Xh + (size_t)(wm + mi * 16 + lh) * EMBED + ks * 32 + lg * 8;
      a[mi].h[0] = *(const v8h*)(xr);
      a[mi].h[1] = *(const v8h*)(xr + 16);
    }
#pragma unroll
    for (int ni = 0; ni < 2; ++ni) {
      const _Float16* wr =
          Wt + (size_t)(wn + ni * 16 + lh) * EMBED + ks * 32 + lg * 8;
      bf[ni].h[0] = *(const v8h*)(wr);
      bf[ni].h[1] = *(const v8h*)(wr + 16);
    }
#pragma unroll
    for (int mi = 0; mi < 4; ++mi)
#pragma unroll
      for (int ni = 0; ni < 2; ++ni)
        c[mi][ni] = wmma_f16(a[mi].v, bf[ni].v, c[mi][ni]);
  }

#pragma unroll
  for (int mi = 0; mi < 4; ++mi)
#pragma unroll
    for (int ni = 0; ni < 2; ++ni) {
      const int n = wn + ni * 16 + lh;
      const float bv = bias[n];
      const int h = n >> 6, d = n & 63;
#pragma unroll
      for (int r = 0; r < 8; ++r) {
        const int m = wm + mi * 16 + lg * 8 + r;
        const int b = m >> 11, s = m & 2047;
        const float val = (c[mi][ni][r] + bv) * scale;
        const size_t oidx =
            vmode ? (((size_t)(b * HEADS + h) * HEAD_DIM + d) * SEQ + s)
                  : (((size_t)(b * HEADS + h) * SEQ + s) * HEAD_DIM + d);
        Out[oidx] = (_Float16)val;
      }
    }
}

// ---------------- stage 3: flash attention with relative bias ----------------
// Q (pre-scaled by 1/8), K: [b,h,s,d] f16.  Vt: [b,h,d,s] f16.
// Out: [b,s,h*64+d] f32.  One wave per 16-query tile; 64-key tiles.
__global__ __launch_bounds__(256) void attn_kernel(
    const _Float16* __restrict__ Q, const _Float16* __restrict__ K,
    const _Float16* __restrict__ Vt, const float* __restrict__ rel_bias,
    float* __restrict__ Out) {
  __shared__ float sbias[NBIAS];
  __shared__ _Float16 sp[8][16 * 64];

  const int tid = threadIdx.x;
  for (int i = tid; i < NBIAS; i += 256) sbias[i] = rel_bias[i];
  __syncthreads();

  const int wave = tid >> 5, lane = tid & 31;
  const int lh = lane & 15, lg = lane >> 4;
  const int bh = blockIdx.x >> 4;          // (b,h) pair
  const int qblk = blockIdx.x & 15;
  const int b = bh >> 4, h = bh & 15;
  const int q0 = qblk * 128 + wave * 16;

  const _Float16* Qb = Q + (size_t)bh * SEQ * HEAD_DIM;
  const _Float16* Kb = K + (size_t)bh * SEQ * HEAD_DIM;
  const _Float16* Vb = Vt + (size_t)bh * HEAD_DIM * SEQ;
  _Float16* pbuf = &sp[wave][0];

  // Q A-fragments for d-steps 0..31 and 32..63, resident all kernel.
  FragAB qa[2];
  {
    const _Float16* qr = Qb + (size_t)(q0 + lh) * HEAD_DIM + lg * 8;
    qa[0].h[0] = *(const v8h*)(qr);
    qa[0].h[1] = *(const v8h*)(qr + 16);
    qa[1].h[0] = *(const v8h*)(qr + 32);
    qa[1].h[1] = *(const v8h*)(qr + 48);
  }

  v8f acc[4];
  float m[8], l[8];
#pragma unroll
  for (int nt = 0; nt < 4; ++nt) acc[nt] = {};
#pragma unroll
  for (int r = 0; r < 8; ++r) { m[r] = -3.0e38f; l[r] = 0.0f; }

  for (int kt = 0; kt < SEQ / 64; ++kt) {
    const int k0 = kt * 64;

    if (kt + 1 < SEQ / 64) {  // prefetch next K/V tiles (global_prefetch_b8)
      __builtin_prefetch(Kb + (size_t)(k0 + 64 + lane) * HEAD_DIM, 0, 1);
      __builtin_prefetch(Kb + (size_t)(k0 + 96 + lane) * HEAD_DIM, 0, 1);
      __builtin_prefetch(Vb + (size_t)lane * SEQ + k0 + 64, 0, 1);
      __builtin_prefetch(Vb + (size_t)(lane + 32) * SEQ + k0 + 64, 0, 1);
    }

    // scores: four 16x16 C tiles (keys k0 + 16*t)
    v8f sc[4];
#pragma unroll
    for (int t = 0; t < 4; ++t) {
      FragAB kb0, kb1;
      const _Float16* kr =
          Kb + (size_t)(k0 + t * 16 + lh) * HEAD_DIM + lg * 8;
      kb0.h[0] = *(const v8h*)(kr);
      kb0.h[1] = *(const v8h*)(kr + 16);
      kb1.h[0] = *(const v8h*)(kr + 32);
      kb1.h[1] = *(const v8h*)(kr + 48);
      v8f s = {};
      s = wmma_f16(qa[0].v, kb0.v, s);
      s = wmma_f16(qa[1].v, kb1.v, s);
      sc[t] = s;
    }

    // relative bias + online softmax (scale already baked into Q)
    float p[4][8];
    const int Mg = q0 + lg * 8;
#pragma unroll
    for (int r = 0; r < 8; ++r) {
      const int qrow = Mg + r;
      float tv[4];
#pragma unroll
      for (int t = 0; t < 4; ++t) {
        int rel = k0 + t * 16 + lh - qrow;
        rel = rel < -MAX_REL ? -MAX_REL : (rel > MAX_REL ? MAX_REL : rel);
        tv[t] = sc[t][r] + sbias[(rel + MAX_REL) * HEADS + h];
      }
      float rmax = fmaxf(fmaxf(tv[0], tv[1]), fmaxf(tv[2], tv[3]));
#pragma unroll
      for (int off = 1; off < 16; off <<= 1)
        rmax = fmaxf(rmax, __shfl_xor(rmax, off, 16));
      const float mn = fmaxf(m[r], rmax);
      const float al = __expf(m[r] - mn);
      m[r] = mn;
      float rs = 0.0f;
#pragma unroll
      for (int t = 0; t < 4; ++t) {
        const float e = __expf(tv[t] - mn);
        p[t][r] = e;
        rs += e;
      }
#pragma unroll
      for (int off = 1; off < 16; off <<= 1)
        rs += __shfl_xor(rs, off, 16);
      l[r] = l[r] * al + rs;
#pragma unroll
      for (int nt = 0; nt < 4; ++nt) acc[nt][r] *= al;
    }

    // stage P (C layout) into LDS, re-read as two A fragments (16x32 each)
#pragma unroll
    for (int r = 0; r < 8; ++r) {
      const int mloc = lg * 8 + r;
#pragma unroll
      for (int t = 0; t < 4; ++t)
        pbuf[mloc * 64 + t * 16 + lh] = (_Float16)p[t][r];
    }
    lds_fence();
    FragAB pa[2];
#pragma unroll
    for (int ksp = 0; ksp < 2; ++ksp) {
      pa[ksp].h[0] = *(const v8h*)(pbuf + lh * 64 + ksp * 32 + lg * 8);
      pa[ksp].h[1] = *(const v8h*)(pbuf + lh * 64 + ksp * 32 + lg * 8 + 16);
    }

    // O += P @ V  (V^T gives contiguous per-lane key runs)
#pragma unroll
    for (int nt = 0; nt < 4; ++nt) {
#pragma unroll
      for (int ksp = 0; ksp < 2; ++ksp) {
        FragAB vb;
        const _Float16* vr =
            Vb + (size_t)(nt * 16 + lh) * SEQ + k0 + ksp * 32 + lg * 8;
        vb.h[0] = *(const v8h*)(vr);
        vb.h[1] = *(const v8h*)(vr + 16);
        acc[nt] = wmma_f16(pa[ksp].v, vb.v, acc[nt]);
      }
    }
  }

  // epilogue: out[b][s][h*64+d] = acc / l
#pragma unroll
  for (int nt = 0; nt < 4; ++nt)
#pragma unroll
    for (int r = 0; r < 8; ++r) {
      const int qrow = q0 + lg * 8 + r;
      const float val = acc[nt][r] / l[r];
      const size_t oi =
          ((size_t)b * SEQ + qrow) * EMBED + h * HEAD_DIM + nt * 16 + lh;
      Out[oi] = val;
    }
}

extern "C" void kernel_launch(void* const* d_in, const int* in_sizes, int n_in,
                              void* d_out, int out_size, void* d_ws,
                              size_t ws_size, hipStream_t stream) {
  (void)in_sizes; (void)n_in; (void)out_size; (void)ws_size;
  const float* x  = (const float*)d_in[0];
  const float* Wq = (const float*)d_in[1];
  const float* bq = (const float*)d_in[2];
  const float* Wk = (const float*)d_in[3];
  const float* bk = (const float*)d_in[4];
  const float* Wv = (const float*)d_in[5];
  const float* bv = (const float*)d_in[6];
  const float* rb = (const float*)d_in[7];

  char* ws = (char*)d_ws;
  const size_t XH_B = (size_t)ROWS * EMBED * 2;       // 16 MB
  const size_t WT_B = (size_t)EMBED * EMBED * 2;      // 2 MB each
  const size_t QKV_B = (size_t)BATCH * HEADS * SEQ * HEAD_DIM * 2;  // 16 MB each
  _Float16* Xh  = (_Float16*)(ws);
  _Float16* Wqt = (_Float16*)(ws + XH_B);
  _Float16* Wkt = (_Float16*)(ws + XH_B + WT_B);
  _Float16* Wvt = (_Float16*)(ws + XH_B + 2 * WT_B);
  _Float16* Qb  = (_Float16*)(ws + XH_B + 3 * WT_B);
  _Float16* Kb  = (_Float16*)(ws + XH_B + 3 * WT_B + QKV_B);
  _Float16* Vtb = (_Float16*)(ws + XH_B + 3 * WT_B + 2 * QKV_B);

  cvt_x_kernel<<<(ROWS * EMBED) / (256 * 4), 256, 0, stream>>>(x, Xh);
  dim3 tb(32, 8), tg(EMBED / 32, EMBED / 32);
  cvt_wT_kernel<<<tg, tb, 0, stream>>>(Wq, Wqt);
  cvt_wT_kernel<<<tg, tb, 0, stream>>>(Wk, Wkt);
  cvt_wT_kernel<<<tg, tb, 0, stream>>>(Wv, Wvt);

  // 64 M-blocks x 8 N-blocks = 512 blocks (block tile 128x128)
  gemm_proj_kernel<<<512, 256, 0, stream>>>(Xh, Wqt, bq, Qb, 0, 0.125f);
  gemm_proj_kernel<<<512, 256, 0, stream>>>(Xh, Wkt, bk, Kb, 0, 1.0f);
  gemm_proj_kernel<<<512, 256, 0, stream>>>(Xh, Wvt, bv, Vtb, 1, 1.0f);

  // 64 (b,h) x 16 query-blocks of 128 rows = 1024 blocks
  attn_kernel<<<1024, 256, 0, stream>>>(Qb, Kb, Vtb, rb, (float*)d_out);
}
